// MixerBlock_833223655539
// MI455X (gfx1250) — matrix operand order repaired and checked by reading, not compiled, not run on hardware
//
#include <hip/hip_runtime.h>

#define NN 50000
#define EE 800000
#define DD 128
#define HH 4
#define DH 32
#define MT (NN / 16)          // 3125 row tiles
#define WAVES_PER_BLOCK 8

typedef __attribute__((ext_vector_type(16))) _Float16 v16h;
typedef __attribute__((ext_vector_type(8)))  float    v8f;
typedef __attribute__((ext_vector_type(4)))  unsigned v4u;
typedef __attribute__((ext_vector_type(4)))  float    v4f;

union ABf16 { v16h h; v4u u[2]; };

// ---- monotone float<->uint mapping for atomic max over signed floats ----
__device__ __forceinline__ unsigned fmap(float f) {
  unsigned b = __float_as_uint(f);
  return (b & 0x80000000u) ? ~b : (b | 0x80000000u);
}
__device__ __forceinline__ float funmap(unsigned u) {
  unsigned b = (u & 0x80000000u) ? (u ^ 0x80000000u) : ~u;
  return __uint_as_float(b);
}

// ---------------- prep: fp32 -> f16 ----------------
__global__ void cvt_x_f16(const float* __restrict__ in, _Float16* __restrict__ out, int n) {
  int i = blockIdx.x * blockDim.x + threadIdx.x;
  if (i < n) out[i] = (_Float16)in[i];
}

__global__ void cvt_w_f16(const float* __restrict__ w0, const float* __restrict__ w1,
                          const float* __restrict__ w2, const float* __restrict__ w3,
                          _Float16* __restrict__ out) {
  int i = blockIdx.x * blockDim.x + threadIdx.x;   // 0..65535 exactly
  int mat = i >> 14;
  int loc = i & 16383;
  const float* w = (mat == 0) ? w0 : (mat == 1) ? w1 : (mat == 2) ? w2 : w3;
  out[i] = (_Float16)w[loc];
}

__global__ void init_stats(unsigned* __restrict__ maxb, float* __restrict__ denom, int n) {
  int i = blockIdx.x * blockDim.x + threadIdx.x;
  if (i < n) { maxb[i] = 0u; denom[i] = 0.0f; }
}

// ---------------- fused Q/K/V/skip projection via WMMA ----------------
// out[m][n] = sum_k x[m][k] * W[n][k] + bias[n]   (x @ W.T + b)
// one wave -> one 16x16 output tile; K loop = 4 x wmma_f32_16x16x32_f16
__global__ void __launch_bounds__(256) qkvs_wmma(
    const _Float16* __restrict__ xh, const _Float16* __restrict__ wh,
    const float* __restrict__ b0, const float* __restrict__ b1,
    const float* __restrict__ b2, const float* __restrict__ b3,
    float* __restrict__ o0, float* __restrict__ o1,
    float* __restrict__ o2, float* __restrict__ o3)
{
  int wave = threadIdx.x >> 5;
  int lane = threadIdx.x & 31;
  int mTile = blockIdx.x * WAVES_PER_BLOCK + wave;
  if (mTile >= MT) return;                 // wave-uniform exit; EXEC stays all-1s for WMMA
  int nTile = blockIdx.y;                  // 0..7
  int mat   = blockIdx.z;                  // 0..3 : q,k,v,skip

  const float* bias = (mat == 0) ? b0 : (mat == 1) ? b1 : (mat == 2) ? b2 : b3;
  float*       out  = (mat == 0) ? o0 : (mat == 1) ? o1 : (mat == 2) ? o2 : o3;
  const _Float16* W = wh + (size_t)mat * (DD * DD);

  int idx = lane & 15;                     // col within tile (B/C/D), row within tile (A)
  int hlf = lane >> 4;                     // lane-half selects K sub-range
  int row = mTile * 16 + idx;
  int col = nTile * 16 + idx;

  // fold bias into accumulator: C/D element (r, idx) -> bias depends on column only
  float bval = bias[col];
  v8f c;
#pragma unroll
  for (int r = 0; r < 8; ++r) c[r] = bval;

  // A layout (16-bit 16x32): lane holds row=idx, K = hlf*8 + {0..7} and +16
  // B layout (16-bit 32x16): lane holds col=idx, K = hlf*8 + {0..7} and +16
  const _Float16* arow = xh + (size_t)row * DD + hlf * 8;
  const _Float16* brow = W  + (size_t)col * DD + hlf * 8;

#pragma unroll
  for (int k0 = 0; k0 < DD; k0 += 32) {
    ABf16 a, b;
    a.u[0] = *reinterpret_cast<const v4u*>(arow + k0);        // K  kb+0..7
    a.u[1] = *reinterpret_cast<const v4u*>(arow + k0 + 16);   // K  kb+16..23
    b.u[0] = *reinterpret_cast<const v4u*>(brow + k0);        // K  kb+0..7
    b.u[1] = *reinterpret_cast<const v4u*>(brow + k0 + 16);   // K  kb+16..23
    c = __builtin_amdgcn_wmma_f32_16x16x32_f16(
            false, a.h, false, b.h, (short)0, c, false, false);
  }

  // D layout: VGPR r -> M = r + hlf*8, N = idx
#pragma unroll
  for (int r = 0; r < 8; ++r) {
    int m = mTile * 16 + hlf * 8 + r;
    out[(size_t)m * DD + col] = c[r];
  }
}

// ---------------- edge phase ----------------
// one thread per (edge, head): alpha = <q[dst,h], k[src,h]> / sqrt(DH), segment max
__global__ void edge_logits(const float* __restrict__ q, const float* __restrict__ k,
                            const int* __restrict__ src, const int* __restrict__ dst,
                            float* __restrict__ alpha, unsigned* __restrict__ maxb, int eh) {
  int gid = blockIdx.x * blockDim.x + threadIdx.x;
  if (gid >= eh) return;
  int e = gid >> 2, h = gid & 3;
  int di = dst[e], si = src[e];
  const float* qp = q + (size_t)di * DD + h * DH;
  const float* kp = k + (size_t)si * DD + h * DH;
  v4f acc = {0.f, 0.f, 0.f, 0.f};
#pragma unroll
  for (int i = 0; i < DH; i += 4) {
    v4f qv = *reinterpret_cast<const v4f*>(qp + i);
    v4f kv = *reinterpret_cast<const v4f*>(kp + i);
    acc += qv * kv;
  }
  float s = (acc[0] + acc[1] + acc[2] + acc[3]) * 0.17677669529663687f; // 1/sqrt(32)
  alpha[gid] = s;
  atomicMax(maxb + (size_t)di * HH + h, fmap(s));
}

__global__ void edge_expsum(float* __restrict__ alpha, const int* __restrict__ dst,
                            const unsigned* __restrict__ maxb, float* __restrict__ denom,
                            int eh) {
  int gid = blockIdx.x * blockDim.x + threadIdx.x;
  if (gid >= eh) return;
  int e = gid >> 2, h = gid & 3;
  int di = dst[e];
  float m = funmap(maxb[(size_t)di * HH + h]);
  float a = __expf(alpha[gid] - m);
  alpha[gid] = a;
  atomicAdd(denom + (size_t)di * HH + h, a);
}

// one thread per (edge, channel): out[dst, c] += (a/denom) * v[src, c]
__global__ void edge_scatter(const float* __restrict__ alpha, const float* __restrict__ denom,
                             const float* __restrict__ v, const int* __restrict__ src,
                             const int* __restrict__ dst, float* __restrict__ out, int ec) {
  int gid = blockIdx.x * blockDim.x + threadIdx.x;
  if (gid >= ec) return;
  int e = gid >> 7;          // / 128
  int c = gid & 127;
  int h = c >> 5;
  int di = dst[e], si = src[e];
  float coef = alpha[(size_t)e * HH + h] / (denom[(size_t)di * HH + h] + 1e-16f);
  atomicAdd(out + (size_t)di * DD + c, coef * v[(size_t)si * DD + c]);
}

__global__ void finalize_residual(const float* __restrict__ x, float* __restrict__ out, int n) {
  int i = blockIdx.x * blockDim.x + threadIdx.x;
  if (i < n) out[i] += x[i];
}

// ---------------- launch ----------------
extern "C" void kernel_launch(void* const* d_in, const int* in_sizes, int n_in,
                              void* d_out, int out_size, void* d_ws, size_t ws_size,
                              hipStream_t stream) {
  (void)in_sizes; (void)n_in; (void)out_size; (void)ws_size;

  const float* x   = (const float*)d_in[0];
  const int*   ei  = (const int*)d_in[1];   // [2, E] int32
  const float* Wq  = (const float*)d_in[2];
  const float* bq  = (const float*)d_in[3];
  const float* Wk  = (const float*)d_in[4];
  const float* bk  = (const float*)d_in[5];
  const float* Wv  = (const float*)d_in[6];
  const float* bv  = (const float*)d_in[7];
  const float* Wsk = (const float*)d_in[8];
  const float* bsk = (const float*)d_in[9];
  float* out = (float*)d_out;

  const int* src = ei;            // edge_index[0]
  const int* dst = ei + EE;       // edge_index[1]

  // workspace carve-out (256B aligned slices)
  char* ws = (char*)d_ws;
  size_t off = 0;
  auto carve = [&](size_t bytes) -> char* {
    char* p = ws + off;
    off = (off + bytes + 255) & ~(size_t)255;
    return p;
  };
  _Float16* xh    = (_Float16*)carve((size_t)NN * DD * sizeof(_Float16));   // 12.8 MB
  _Float16* wh    = (_Float16*)carve((size_t)4 * DD * DD * sizeof(_Float16));
  float*    qbuf  = (float*)carve((size_t)NN * DD * sizeof(float));          // 25.6 MB
  float*    kbuf  = (float*)carve((size_t)NN * DD * sizeof(float));
  float*    vbuf  = (float*)carve((size_t)NN * DD * sizeof(float));
  float*    alpha = (float*)carve((size_t)EE * HH * sizeof(float));          // 12.8 MB
  unsigned* maxb  = (unsigned*)carve((size_t)NN * HH * sizeof(unsigned));
  float*    denom = (float*)carve((size_t)NN * HH * sizeof(float));

  const int nd = NN * DD;          // 6.4M
  const int eh = EE * HH;          // 3.2M
  const int ec = EE * DD;          // 102.4M
  const int nh = NN * HH;          // 200K

  cvt_x_f16<<<(nd + 255) / 256, 256, 0, stream>>>(x, xh, nd);
  cvt_w_f16<<<(4 * DD * DD) / 256, 256, 0, stream>>>(Wq, Wk, Wv, Wsk, wh);
  init_stats<<<(nh + 255) / 256, 256, 0, stream>>>(maxb, denom, nh);

  // skip projection writes d_out directly (every element overwritten each call)
  dim3 ggrid((MT + WAVES_PER_BLOCK - 1) / WAVES_PER_BLOCK, DD / 16, 4);
  qkvs_wmma<<<ggrid, 256, 0, stream>>>(xh, wh, bq, bk, bv, bsk, qbuf, kbuf, vbuf, out);

  edge_logits<<<(eh + 255) / 256, 256, 0, stream>>>(qbuf, kbuf, src, dst, alpha, maxb, eh);
  edge_expsum<<<(eh + 255) / 256, 256, 0, stream>>>(alpha, dst, maxb, denom, eh);
  edge_scatter<<<(ec + 255) / 256, 256, 0, stream>>>(alpha, denom, vbuf, src, dst, out, ec);
  finalize_residual<<<(nd + 255) / 256, 256, 0, stream>>>(x, out, nd);
}
